// AttnLayer_44951127719954
// MI455X (gfx1250) — matrix operand order repaired and checked by reading, not compiled, not run on hardware
//
#include <hip/hip_runtime.h>

// CDNA5 / gfx1250 random-key attention.
// B=2, L=4096, NH=8, H=64, N=64 gathered keys. K/V fit in L2 (16MB each vs
// 192MB) -> L2 gather-bandwidth bound (~2GB gathered reads total). One WG per
// query: stage 64 gathered contiguous 2KB rows into LDS, all 8 heads (one
// wave each) consume them via v_wmma_f32_16x16x32_f16.
//
// Gather path: prefer CDNA5 async global->LDS DMA (no VGPR round-trip,
// tracked by ASYNCcnt); fall back to float4 loads + f16 convert if the
// builtin is unavailable in this toolchain.

typedef __attribute__((ext_vector_type(16))) _Float16 v16h;
typedef __attribute__((ext_vector_type(8)))  float    v8f;
typedef __attribute__((ext_vector_type(4)))  int      v4i;

constexpr int Lq    = 4096;     // sequence length
constexpr int NH    = 8;        // heads
constexpr int Hd    = 64;       // head dim
constexpr int NKEYS = 64;       // gathered keys per query
constexpr int HD    = NH * Hd;  // 512 floats per row of k/v

#if defined(__HIP_DEVICE_COMPILE__) && \
    __has_builtin(__builtin_amdgcn_global_load_async_to_lds_b128)
#define USE_ASYNC_LDS 1
typedef float lds_t;            // DMA can't convert: stage f32 (128KB, 2 WG/WGP)
typedef __attribute__((address_space(1))) v4i g_v4i;
typedef __attribute__((address_space(3))) v4i l_v4i;
#else
#define USE_ASYNC_LDS 0
typedef _Float16 lds_t;         // VGPR-staged path converts to f16 (64KB)
#endif

__device__ __forceinline__ void async_wait_zero() {
#if USE_ASYNC_LDS
#if __has_builtin(__builtin_amdgcn_s_wait_asynccnt)
  __builtin_amdgcn_s_wait_asynccnt(0);
#else
  asm volatile("s_wait_asynccnt 0x0" ::: "memory");
#endif
#endif
}

// Gather NKEYS contiguous 2KB rows (row = b*Lq + idxs[r]) from `src` into buf.
__device__ __forceinline__ void gather_rows(const float* __restrict__ src,
                                            const int* __restrict__ idxs,
                                            int b, lds_t (*buf)[HD], int tid) {
  for (int i = tid; i < NKEYS * (HD / 4); i += 256) {
    const int row = i >> 7;             // i / 128 float4-chunks per row
    const int c4  = i & 127;
    const float* g = src + ((size_t)b * Lq + idxs[row]) * HD + c4 * 4;
#if USE_ASYNC_LDS
    __builtin_amdgcn_global_load_async_to_lds_b128(
        (g_v4i*)(void*)g, (l_v4i*)(void*)&buf[row][c4 * 4], 0, 0);
#else
    const float4 d = *(const float4*)g;
    _Float16* dst = &buf[row][c4 * 4];
    dst[0] = (_Float16)d.x; dst[1] = (_Float16)d.y;
    dst[2] = (_Float16)d.z; dst[3] = (_Float16)d.w;
#endif
  }
  async_wait_zero();                    // drain this wave's ASYNCcnt
}

__global__ __launch_bounds__(256) void rand_attn_kernel(
    const float* __restrict__ q, const float* __restrict__ k,
    const float* __restrict__ v, const int* __restrict__ sel,
    float* __restrict__ out)
{
  // one buffer, staged K then V -> better occupancy than K+V resident
  __shared__ lds_t buf[NKEYS][HD];
  __shared__ float wbuf[NH][NKEYS];     // per-head softmax weights
  __shared__ int   idxs[NKEYS];

  const int bq   = blockIdx.x;          // flat (b, q) index, 0..B*L-1
  const int b    = bq / Lq;
  const int tid  = threadIdx.x;
  const int lane = tid & 31;            // wave32
  const int head = tid >> 5;            // wave id == head id (8 waves)
  const int col  = lane & 15;
  const int hi   = lane >> 4;           // half-wave selector (ISA frag layout)

  if (tid < NKEYS) idxs[tid] = sel[(size_t)bq * NKEYS + tid];
  __syncthreads();

  // ---------------- gather K rows -> LDS -----------------------------------
  gather_rows(k, idxs, b, buf, tid);

  // q fragment for this head, scaled by H^-0.5, replicated into all 16 B-cols.
  // B(32x16) layout: lane holds column N=col; element e -> K = 8*hi + e + (e & 8).
  v16h qf[2];
  {
    const float scale = 0.125f;         // 64^-0.5
    const float* qp = q + (size_t)bq * HD + head * Hd;
#pragma unroll
    for (int c = 0; c < 2; ++c)
#pragma unroll
      for (int e = 0; e < 16; ++e)
        qf[c][e] = (_Float16)(qp[c * 32 + 8 * hi + e + (e & 8)] * scale);
  }
  __syncthreads();

  // ---------------- QK^T: 64 scores per head via 8 WMMAs --------------------
  // A(16x32) layout: lane holds row M=col; element e=(p<<1|bb) ->
  //   K = (p<4 ? 0 : 16) + 8*hi + 2*(p&3) + bb
  float sc[4][8];
#pragma unroll
  for (int t = 0; t < 4; ++t) {         // key tile: keys t*16 .. t*16+15
    v8f acc = {};
#pragma unroll
    for (int c = 0; c < 2; ++c) {       // contraction chunk over head dim
      v16h af;
      const int keyrow = t * 16 + col;
#pragma unroll
      for (int e = 0; e < 16; ++e) {
        const int p = e >> 1, bb = e & 1;
        const int kk = ((p < 4) ? 0 : 16) + 8 * hi + ((p & 3) * 2 + bb);
        af[e] = (_Float16)buf[keyrow][head * Hd + c * 32 + kk];
      }
      acc = __builtin_amdgcn_wmma_f32_16x16x32_f16(
          false, af, false, qf[c], (short)0, acc, false, false);
    }
#pragma unroll
    for (int r = 0; r < 8; ++r) sc[t][r] = acc[r];
  }

  // lane holds scores for keys m = t*16 + 8*hi + r (replicated over 16 cols).
  // softmax over all 64 keys: local reduce + cross-half shfl_xor(16).
  float mx = -1e30f;
#pragma unroll
  for (int t = 0; t < 4; ++t)
#pragma unroll
    for (int r = 0; r < 8; ++r) mx = fmaxf(mx, sc[t][r]);
  mx = fmaxf(mx, __shfl_xor(mx, 16, 32));
  float sum = 0.f;
#pragma unroll
  for (int t = 0; t < 4; ++t)
#pragma unroll
    for (int r = 0; r < 8; ++r) {
      const float e = __expf(sc[t][r] - mx);
      sc[t][r] = e;
      sum += e;
    }
  sum += __shfl_xor(sum, 16, 32);
  const float inv = 1.f / sum;
  if (col == 0) {                       // one writer per (head, key)
#pragma unroll
    for (int t = 0; t < 4; ++t)
#pragma unroll
      for (int r = 0; r < 8; ++r)
        wbuf[head][t * 16 + 8 * hi + r] = sc[t][r] * inv;
  }
  __syncthreads();                      // everyone done with K in buf

  // ---------------- gather V rows -> same LDS buffer ------------------------
  gather_rows(v, idxs, b, buf, tid);
  __syncthreads();

  // ---------------- A·V: z[64] = sum_n w[n] * V[n, head, :] -----------------
  // A2 = weights replicated across all 16 rows (lane's row = col, data is
  // row-independent); contraction = keys, 2 chunks of 32.
  v16h wf[2];
#pragma unroll
  for (int c = 0; c < 2; ++c)
#pragma unroll
    for (int e = 0; e < 16; ++e) {
      const int p = e >> 1, bb = e & 1;
      const int kk = ((p < 4) ? 0 : 16) + 8 * hi + ((p & 3) * 2 + bb);
      wf[c][e] = (_Float16)wbuf[head][c * 32 + kk];
    }

  float* outp = out + (size_t)bq * HD + head * Hd;
#pragma unroll
  for (int t = 0; t < 4; ++t) {         // 16 output dims per tile
    v8f acc = {};
#pragma unroll
    for (int c = 0; c < 2; ++c) {
      v16h bf;                          // B(32x16): V[key, head, t*16+col]
#pragma unroll
      for (int e = 0; e < 16; ++e)
        bf[e] = (_Float16)buf[c * 32 + 8 * hi + e + (e & 8)][head * Hd + t * 16 + col];
      acc = __builtin_amdgcn_wmma_f32_16x16x32_f16(
          false, wf[c], false, bf, (short)0, acc, false, false);
    }
    if (lane < 16) outp[t * 16 + col] = acc[0];  // rows identical; take M=0
  }
}

extern "C" void kernel_launch(void* const* d_in, const int* in_sizes, int n_in,
                              void* d_out, int out_size, void* d_ws, size_t ws_size,
                              hipStream_t stream) {
  const float* q   = (const float*)d_in[0];
  const float* k   = (const float*)d_in[1];
  const float* v   = (const float*)d_in[2];
  const int*   sel = (const int*)d_in[3];
  float* out = (float*)d_out;

  const int nBQ = in_sizes[3] / NKEYS;  // B*L = 8192 query positions
  rand_attn_kernel<<<nBQ, 256, 0, stream>>>(q, k, v, sel, out);
}
